// WordnestMoE_16226386444623
// MI455X (gfx1250) — compile-verified, matrix operand order checked by reading, not run on hardware
//
#include <hip/hip_runtime.h>
#include <hip/hip_bf16.h>
#include <stdint.h>

// ---------------------------------------------------------------------------
// MoE FFN for MI455X (gfx1250): bf16 WMMA + async-to-LDS double buffering.
// Routed top-2 experts (4x less FLOPs than the dense reference loop).
// ---------------------------------------------------------------------------

#define BM 128
#define BN 128
#define BK 32
#define LDT 40   // LDS row stride in ushorts (32 + 8 pad -> 80B, conflict-free)
#define BUFE ((BM + BN) * LDT)   // ushorts per double-buffer stage

typedef __attribute__((ext_vector_type(16))) __bf16 v16bf;
typedef __attribute__((ext_vector_type(8)))  float  v8f;

struct U128 { unsigned int x0, x1, x2, x3; };

union Frag16 {
  v16bf v;
  U128  q[2];
};

__device__ __forceinline__ unsigned short f32_to_bf16(float f) {
  unsigned int u = __float_as_uint(f);
  u = u + 0x7FFFu + ((u >> 16) & 1u);   // round-to-nearest-even
  return (unsigned short)(u >> 16);
}

// 32-bit LDS byte offset of a pointer into a __shared__ array.
__device__ __forceinline__ unsigned lds_off_of(const unsigned short* p) {
  return (unsigned)(unsigned long long)
         (__attribute__((address_space(3))) const unsigned short*)p;
}

// Async DMA: 16B global -> LDS, per-lane addresses, tracked by ASYNCcnt.
__device__ __forceinline__ void async_ld_b128(unsigned lds_off, const void* g) {
  asm volatile("global_load_async_to_lds_b128 %0, %1, off"
               :: "v"(lds_off), "v"(g)
               : "memory");
}

__device__ __forceinline__ void wait_asynccnt0() {
#if __has_builtin(__builtin_amdgcn_s_wait_asynccnt)
  __builtin_amdgcn_s_wait_asynccnt(0);
#else
  asm volatile("s_wait_asynccnt 0x0" ::: "memory");
#endif
}

// Per-lane WMMA fragment from a row-major LDS tile (row = M for A, N for B^T).
// 16-bit operand layout (ISA 7.12.2): lanes 0-15 take K 0-7 (+16), lanes 16-31
// take K 8-15 (+24); element row = lane % 16.
__device__ __forceinline__ v16bf load_frag_lds(const unsigned short* tileBase, int row) {
  const int lane = threadIdx.x & 31;
  const int r    = row + (lane & 15);
  const int half = (lane >> 4) * 8;
  const unsigned short* p = tileBase + r * LDT + half;
  Frag16 f;
  f.q[0] = *(const U128*)(p);        // K = half .. half+7
  f.q[1] = *(const U128*)(p + 16);   // K = half+16 .. half+23
  return f.v;
}

// ------------------------------ conversion ---------------------------------
__global__ __launch_bounds__(256)
void cvt_f32_bf16(const float* __restrict__ src, unsigned short* __restrict__ dst,
                  long long n) {
  long long i = (long long)blockIdx.x * blockDim.x + threadIdx.x;
  const long long stride = (long long)gridDim.x * blockDim.x;
  for (; i < n; i += stride) dst[i] = f32_to_bf16(src[i]);
}

// Tiled transpose + convert: src [R,C] f32 row-major -> dst [C,R] bf16.
// (Weights go in transposed so GEMM B-tiles are contiguous 16B row chunks.)
__global__ __launch_bounds__(256)
void cvt_transpose_f32_bf16(const float* __restrict__ src,
                            unsigned short* __restrict__ dst,
                            int R, int C) {
  __shared__ unsigned short tile[32][33];
  const size_t mstride = (size_t)R * C;
  src += (size_t)blockIdx.z * mstride;
  dst += (size_t)blockIdx.z * mstride;
  const int bx = blockIdx.x * 32;   // col base in src
  const int by = blockIdx.y * 32;   // row base in src
  const int tx = threadIdx.x & 31;
  const int ty = threadIdx.x >> 5;  // 0..7
#pragma unroll
  for (int i = 0; i < 4; ++i) {
    const int r = by + ty + i * 8;
    tile[ty + i * 8][tx] = f32_to_bf16(src[(size_t)r * C + bx + tx]);
  }
  __syncthreads();
#pragma unroll
  for (int i = 0; i < 4; ++i) {
    const int c = bx + ty + i * 8;          // output row = src col
    dst[(size_t)c * R + by + tx] = tile[tx][ty + i * 8];
  }
}

// ------------------------------- gating ------------------------------------
__global__ __launch_bounds__(256)
void gate_kernel(const float* __restrict__ X, const float* __restrict__ Wg,
                 const float* __restrict__ bg, const float* __restrict__ bias,
                 int* __restrict__ eTop, float* __restrict__ wTop,
                 int* __restrict__ counts, int Ntok, int D) {
  const int wid  = (int)((blockIdx.x * blockDim.x + threadIdx.x) >> 5);
  const int lane = threadIdx.x & 31;
  if (wid >= Ntok) return;
  const float* xr = X + (size_t)wid * D;
  float acc[8] = {0.f, 0.f, 0.f, 0.f, 0.f, 0.f, 0.f, 0.f};
  for (int d = lane; d < D; d += 32) {
    const float xv = xr[d];
    const float* wr = Wg + (size_t)d * 8;
#pragma unroll
    for (int e = 0; e < 8; ++e) acc[e] += xv * wr[e];
  }
#pragma unroll
  for (int e = 0; e < 8; ++e) {
#pragma unroll
    for (int off = 16; off > 0; off >>= 1) acc[e] += __shfl_xor(acc[e], off, 32);
  }
  if (lane == 0) {
    float s[8];
#pragma unroll
    for (int e = 0; e < 8; ++e) {
      const float logit = acc[e] + bg[e] + bias[e];
      s[e] = 1.0f / (1.0f + __expf(-logit));
    }
    int b0 = 0;
#pragma unroll
    for (int e = 1; e < 8; ++e) if (s[e] > s[b0]) b0 = e;
    int b1 = (b0 == 0) ? 1 : 0;
#pragma unroll
    for (int e = 0; e < 8; ++e) if (e != b0 && s[e] > s[b1]) b1 = e;
    const float e1 = __expf(s[b1] - s[b0]);   // softmax over {s[b0], s[b1]}
    const float w0 = 1.0f / (1.0f + e1);
    const float w1 = e1 * w0;
    eTop[wid * 2 + 0] = b0;  eTop[wid * 2 + 1] = b1;
    wTop[wid * 2 + 0] = w0;  wTop[wid * 2 + 1] = w1;
    atomicAdd(&counts[b0], 1);
    atomicAdd(&counts[b1], 1);
  }
}

__global__ void offsets_kernel(const int* __restrict__ counts, int* __restrict__ offs,
                               int* __restrict__ cursor, int E) {
  if (threadIdx.x == 0 && blockIdx.x == 0) {
    int o = 0;
    for (int e = 0; e < E; ++e) { offs[e] = o; cursor[e] = o; o += counts[e]; }
  }
}

__global__ __launch_bounds__(256)
void fill_kernel(const int* __restrict__ eTop, const float* __restrict__ wTop,
                 int* __restrict__ cursor, int* __restrict__ rowTok,
                 float* __restrict__ rowGate, int Ntok) {
  const int t = blockIdx.x * blockDim.x + threadIdx.x;
  if (t >= Ntok) return;
#pragma unroll
  for (int k = 0; k < 2; ++k) {
    const int e = eTop[t * 2 + k];
    const int pos = atomicAdd(&cursor[e], 1);   // cursor starts at offs[e]
    rowTok[pos]  = t;
    rowGate[pos] = wTop[t * 2 + k];
  }
}

// --------------------------- FFN layer 1 GEMM ------------------------------
// H = silu(A @ W1 + b1); A rows gathered via rowTok (expert) or dense.
// Double-buffered async-to-LDS staging; 8 waves x (2x4) wmma accumulators.
__global__ __launch_bounds__(256, 2)
void ffn1_gemm(const unsigned short* __restrict__ Xb,     // [Ntok, Dk] bf16
               const unsigned short* __restrict__ W1t,    // [(E,) Nn, Dk] bf16 (transposed)
               const float* __restrict__ B1all,           // [(E,) Nn]
               unsigned short* __restrict__ Hout,         // [rows, Nn] bf16
               const int* __restrict__ rowTok,            // null => dense
               const int* __restrict__ segOff,
               const int* __restrict__ segCnt,
               int Mdense, int Dk, int Nn) {
  __shared__ __align__(16) unsigned short lds[2 * BUFE];

  const int e = blockIdx.z;
  const unsigned short* W = W1t;
  const float* bias = B1all;
  int rowsStart = 0, rowsCnt = Mdense;
  if (rowTok) {
    W         = W1t  + (size_t)e * Dk * Nn;
    bias      = B1all + (size_t)e * Nn;
    rowsStart = segOff[e];
    rowsCnt   = segCnt[e];
  }
  const int m0 = blockIdx.y * BM;
  if (m0 >= rowsCnt) return;
  const int n0 = blockIdx.x * BN;

  const int tid  = threadIdx.x;
  const int lane = tid & 31;
  const int wave = tid >> 5;
  const int wm = (wave & 3) * 32;
  const int wn = (wave >> 2) * 64;

  // Each thread owns 2 A-vectors and 2 B-vectors (16B each) per tile.
  int sRow[2], sCol[2], aTok[2];
  const unsigned short* gA[2];
  const unsigned short* gB[2];
  unsigned aDst[2], bDst[2];
#pragma unroll
  for (int l = 0; l < 2; ++l) {
    const int v = tid + l * 256;        // 512 vectors per tile
    sRow[l] = v >> 2;
    sCol[l] = (v & 3) * 8;
    const int r = m0 + sRow[l];
    aTok[l] = (r < rowsCnt) ? (rowTok ? rowTok[rowsStart + r] : r) : -1;
    gA[l] = (aTok[l] >= 0) ? (Xb + (size_t)aTok[l] * Dk + sCol[l]) : nullptr;
    gB[l] = W + (size_t)(n0 + sRow[l]) * Dk + sCol[l];
    aDst[l] = lds_off_of(&lds[sRow[l] * LDT + sCol[l]]);
    bDst[l] = lds_off_of(&lds[BM * LDT + sRow[l] * LDT + sCol[l]]);
  }
  // Zero invalid A chunks once (never async-written afterwards).
  const U128 z128 = {0u, 0u, 0u, 0u};
#pragma unroll
  for (int l = 0; l < 2; ++l)
    if (aTok[l] < 0) {
      *(U128*)&lds[sRow[l] * LDT + sCol[l]]        = z128;
      *(U128*)&lds[BUFE + sRow[l] * LDT + sCol[l]] = z128;
    }

  const unsigned bufBytes = BUFE * 2u;
  auto issue = [&](int buf) {
    const unsigned boff = buf ? bufBytes : 0u;
#pragma unroll
    for (int l = 0; l < 2; ++l) {
      if (aTok[l] >= 0) { async_ld_b128(aDst[l] + boff, gA[l]); gA[l] += BK; }
      async_ld_b128(bDst[l] + boff, gB[l]); gB[l] += BK;
    }
  };

  const v8f zero8 = {0.f, 0.f, 0.f, 0.f, 0.f, 0.f, 0.f, 0.f};
  v8f acc[2][4];
#pragma unroll
  for (int i = 0; i < 2; ++i)
#pragma unroll
    for (int j = 0; j < 4; ++j) acc[i][j] = zero8;

  issue(0);
  int cur = 0;
  const int steps = Dk / BK;
  for (int s = 0; s < steps; ++s) {
    wait_asynccnt0();
    __syncthreads();                 // async data visible; prev-buf reads done
    if (s + 1 < steps) issue(cur ^ 1);
    const unsigned short* Abase = &lds[cur * BUFE];
    const unsigned short* Bbase = Abase + BM * LDT;
    v16bf aF[2], bF[4];
#pragma unroll
    for (int i = 0; i < 2; ++i) aF[i] = load_frag_lds(Abase, wm + i * 16);
#pragma unroll
    for (int j = 0; j < 4; ++j) bF[j] = load_frag_lds(Bbase, wn + j * 16);
#pragma unroll
    for (int i = 0; i < 2; ++i)
#pragma unroll
      for (int j = 0; j < 4; ++j)
        acc[i][j] = __builtin_amdgcn_wmma_f32_16x16x32_bf16(
            false, aF[i], false, bF[j], (short)0, acc[i][j], false, false);
    cur ^= 1;
  }

  const int colLane = lane & 15;
  const int rowHalf = (lane >> 4) * 8;
#pragma unroll
  for (int i = 0; i < 2; ++i) {
#pragma unroll
    for (int j = 0; j < 4; ++j) {
      const int col = n0 + wn + j * 16 + colLane;
      const float bb = bias[col];
#pragma unroll
      for (int r = 0; r < 8; ++r) {
        const int gr = m0 + wm + i * 16 + rowHalf + r;
        if (gr < rowsCnt) {
          const float xv = acc[i][j][r] + bb;
          const float sv = xv / (1.0f + __expf(-xv));   // SiLU
          Hout[(size_t)(rowsStart + gr) * Nn + col] = f32_to_bf16(sv);
        }
      }
    }
  }
}

// --------------------------- FFN layer 2 GEMM ------------------------------
// Dense mode:  Out = Xres + H @ W2 + b2   (writes every element of d_out)
// Expert mode: Out[tok] += gate * (H @ W2 + b2)   via atomicAdd scatter
__global__ __launch_bounds__(256, 2)
void ffn2_gemm(const unsigned short* __restrict__ Hbuf,   // [rows(+pad), Dk] bf16
               const unsigned short* __restrict__ W2t,    // [(E,) Nn, Dk] bf16 (transposed)
               const float* __restrict__ B2all,           // [(E,) Nn]
               const float* __restrict__ Xres,            // dense mode only
               float* __restrict__ Out,                   // [Mdense, Nn]
               const int* __restrict__ rowTok,
               const float* __restrict__ rowGate,
               const int* __restrict__ segOff,
               const int* __restrict__ segCnt,
               int Mdense, int Dk, int Nn) {
  __shared__ __align__(16) unsigned short lds[2 * BUFE];

  const int e = blockIdx.z;
  const unsigned short* W = W2t;
  const float* bias = B2all;
  int rowsStart = 0, rowsCnt = Mdense;
  if (rowTok) {
    W         = W2t  + (size_t)e * Dk * Nn;
    bias      = B2all + (size_t)e * Nn;
    rowsStart = segOff[e];
    rowsCnt   = segCnt[e];
  }
  const int m0 = blockIdx.y * BM;
  if (m0 >= rowsCnt) return;
  const int n0 = blockIdx.x * BN;

  const int tid  = threadIdx.x;
  const int lane = tid & 31;
  const int wave = tid >> 5;
  const int wm = (wave & 3) * 32;
  const int wn = (wave >> 2) * 64;

  int sRow[2], sCol[2];
  const unsigned short* gA[2];
  const unsigned short* gB[2];
  unsigned aDst[2], bDst[2];
#pragma unroll
  for (int l = 0; l < 2; ++l) {
    const int v = tid + l * 256;
    sRow[l] = v >> 2;
    sCol[l] = (v & 3) * 8;
    gA[l] = Hbuf + (size_t)(rowsStart + m0 + sRow[l]) * Dk + sCol[l]; // padded
    gB[l] = W + (size_t)(n0 + sRow[l]) * Dk + sCol[l];
    aDst[l] = lds_off_of(&lds[sRow[l] * LDT + sCol[l]]);
    bDst[l] = lds_off_of(&lds[BM * LDT + sRow[l] * LDT + sCol[l]]);
  }
  const unsigned bufBytes = BUFE * 2u;
  auto issue = [&](int buf) {
    const unsigned boff = buf ? bufBytes : 0u;
#pragma unroll
    for (int l = 0; l < 2; ++l) {
      async_ld_b128(aDst[l] + boff, gA[l]); gA[l] += BK;
      async_ld_b128(bDst[l] + boff, gB[l]); gB[l] += BK;
    }
  };

  const v8f zero8 = {0.f, 0.f, 0.f, 0.f, 0.f, 0.f, 0.f, 0.f};
  v8f acc[2][4];
#pragma unroll
  for (int i = 0; i < 2; ++i)
#pragma unroll
    for (int j = 0; j < 4; ++j) acc[i][j] = zero8;

  issue(0);
  int cur = 0;
  const int steps = Dk / BK;
  for (int s = 0; s < steps; ++s) {
    wait_asynccnt0();
    __syncthreads();
    if (s + 1 < steps) issue(cur ^ 1);
    const unsigned short* Abase = &lds[cur * BUFE];
    const unsigned short* Bbase = Abase + BM * LDT;
    v16bf aF[2], bF[4];
#pragma unroll
    for (int i = 0; i < 2; ++i) aF[i] = load_frag_lds(Abase, wm + i * 16);
#pragma unroll
    for (int j = 0; j < 4; ++j) bF[j] = load_frag_lds(Bbase, wn + j * 16);
#pragma unroll
    for (int i = 0; i < 2; ++i)
#pragma unroll
      for (int j = 0; j < 4; ++j)
        acc[i][j] = __builtin_amdgcn_wmma_f32_16x16x32_bf16(
            false, aF[i], false, bF[j], (short)0, acc[i][j], false, false);
    cur ^= 1;
  }

  const int colLane = lane & 15;
  const int rowHalf = (lane >> 4) * 8;
#pragma unroll
  for (int i = 0; i < 2; ++i) {
#pragma unroll
    for (int j = 0; j < 4; ++j) {
      const int col = n0 + wn + j * 16 + colLane;
      const float bb = bias[col];
#pragma unroll
      for (int r = 0; r < 8; ++r) {
        const int gr = m0 + wm + i * 16 + rowHalf + r;
        if (gr < rowsCnt) {
          const float v = acc[i][j][r] + bb;
          if (rowTok) {
            const int   tok = rowTok[rowsStart + gr];
            const float g   = rowGate[rowsStart + gr];
            atomicAdd(&Out[(size_t)tok * Nn + col], g * v);
          } else {
            const size_t idx = (size_t)gr * Nn + col;
            Out[idx] = Xres[idx] + v;   // residual fused
          }
        }
      }
    }
  }
}

// ------------------------------- launcher ----------------------------------
extern "C" void kernel_launch(void* const* d_in, const int* in_sizes, int n_in,
                              void* d_out, int out_size, void* d_ws, size_t ws_size,
                              hipStream_t stream) {
  (void)in_sizes; (void)n_in; (void)out_size; (void)ws_size;
  const int D = 1024, H = 4096, E = 8;
  const int N = 4 * 2048;          // B*T tokens
  const int TOTAL = N * 2;         // top-2 assignments

  const float* x    = (const float*)d_in[0];
  const float* Ws1  = (const float*)d_in[1];
  const float* bs1  = (const float*)d_in[2];
  const float* Ws2  = (const float*)d_in[3];
  const float* bs2  = (const float*)d_in[4];
  const float* We1  = (const float*)d_in[5];
  const float* be1  = (const float*)d_in[6];
  const float* We2  = (const float*)d_in[7];
  const float* be2  = (const float*)d_in[8];
  const float* Wg   = (const float*)d_in[9];
  const float* bg   = (const float*)d_in[10];
  const float* bias = (const float*)d_in[11];
  float* out = (float*)d_out;

  uint8_t* ws = (uint8_t*)d_ws;
  size_t off = 0;
  auto take = [&](size_t bytes) -> void* {
    off = (off + 255) & ~(size_t)255;
    void* p = ws + off;
    off += bytes;
    return p;
  };
  unsigned short* xb   = (unsigned short*)take((size_t)N * D * 2);
  unsigned short* ws1t = (unsigned short*)take((size_t)D * H * 2);       // [H,D]
  unsigned short* ws2t = (unsigned short*)take((size_t)H * D * 2);       // [D,H]
  unsigned short* we1t = (unsigned short*)take((size_t)E * D * H * 2);   // [E,H,D]
  unsigned short* we2t = (unsigned short*)take((size_t)E * H * D * 2);   // [E,D,H]
  unsigned short* hbuf = (unsigned short*)take(((size_t)TOTAL + 128) * H * 2);
  int*   rowTok  = (int*)take((size_t)TOTAL * 4);
  float* rowGate = (float*)take((size_t)TOTAL * 4);
  int*   eTop    = (int*)take((size_t)N * 2 * 4);
  float* wTop    = (float*)take((size_t)N * 2 * 4);
  int*   counts  = (int*)take(64);
  int*   offs    = (int*)take(64);
  int*   cursor  = (int*)take(64);

  hipMemsetAsync(counts, 0, E * sizeof(int), stream);

  // fp32 -> bf16; weights transposed so GEMM B-tiles are row-contiguous.
  cvt_f32_bf16<<<2048, 256, 0, stream>>>(x, xb, (long long)N * D);
  {
    dim3 t1(H / 32, D / 32, 1);   // Ws1 [D,H] -> [H,D]
    cvt_transpose_f32_bf16<<<t1, 256, 0, stream>>>(Ws1, ws1t, D, H);
    dim3 t2(D / 32, H / 32, 1);   // Ws2 [H,D] -> [D,H]
    cvt_transpose_f32_bf16<<<t2, 256, 0, stream>>>(Ws2, ws2t, H, D);
    dim3 t3(H / 32, D / 32, E);   // We1 [E,D,H] -> [E,H,D]
    cvt_transpose_f32_bf16<<<t3, 256, 0, stream>>>(We1, we1t, D, H);
    dim3 t4(D / 32, H / 32, E);   // We2 [E,H,D] -> [E,D,H]
    cvt_transpose_f32_bf16<<<t4, 256, 0, stream>>>(We2, we2t, H, D);
  }

  // gating + routing tables
  gate_kernel<<<N / 8, 256, 0, stream>>>(x, Wg, bg, bias, eTop, wTop, counts, N, D);
  offsets_kernel<<<1, 32, 0, stream>>>(counts, offs, cursor, E);
  fill_kernel<<<N / 256, 256, 0, stream>>>(eTop, wTop, cursor, rowTok, rowGate, N);

  // shared FFN: h = silu(x@Ws1+b1); out = x + h@Ws2 + b2
  {
    dim3 g1(H / BN, N / BM, 1);
    ffn1_gemm<<<g1, 256, 0, stream>>>(xb, ws1t, bs1, hbuf,
                                      nullptr, nullptr, nullptr, N, D, H);
    dim3 g2(D / BN, N / BM, 1);
    ffn2_gemm<<<g2, 256, 0, stream>>>(hbuf, ws2t, bs2, x, out,
                                      nullptr, nullptr, nullptr, nullptr, N, H, D);
  }
  // routed experts: gather rows, per-expert GEMMs, gated atomic scatter-add
  {
    dim3 g3(H / BN, N / BM, E);   // worst-case tiles; blocks early-exit on count
    ffn1_gemm<<<g3, 256, 0, stream>>>(xb, we1t, be1, hbuf,
                                      rowTok, offs, counts, N, D, H);
    dim3 g4(D / BN, N / BM, E);
    ffn2_gemm<<<g4, 256, 0, stream>>>(hbuf, we2t, be2, nullptr, out,
                                      rowTok, rowGate, offs, counts, N, H, D);
  }
}
